// AASIST_31112743092575
// MI455X (gfx1250) — matrix operand order repaired and checked by reading, not compiled, not run on hardware
//
#include <hip/hip_runtime.h>
#include <math.h>

typedef __attribute__((ext_vector_type(16))) _Float16 v16h;
typedef __attribute__((ext_vector_type(8)))  float    v8f;

#define TEMP_INV    0.5f
#define BN_EPS_C    1e-5f
#define SELU_SCALE  1.0507009873554805f
#define SELU_ALPHA  1.6732632423543772f

#define BB   4
#define NN   512
#define DD   64
#define OO   64
#define NKP  256          // nodes kept by GraphPool (512 * 0.5)
#define XSTR 72           // padded LDS row stride (f16 elems) to dodge bank conflicts

// Branchless tanh: hardware V_TANH_F32 on gfx1250, fast-exp fallback otherwise.
__device__ __forceinline__ float fast_tanh(float x) {
#if __has_builtin(__builtin_amdgcn_tanhf)
    return __builtin_amdgcn_tanhf(x);
#else
    float e = __expf(2.0f * x);
    return (e - 1.0f) * __builtin_amdgcn_rcpf(e + 1.0f);
#endif
}

// ---------------------------------------------------------------------------
// Kernel 1: fused GraphAttentionLayer.
// grid = B * (N/8) blocks of 256 threads (8 waves); wave w owns row i = ib + w.
// Per (b,i): Z[j,o] = sum_d (x_i[d]*x_j[d]) * W[d,o]  via v_wmma_f32_16x16x32_f16,
// then logits[j] = (1/T) * sum_o v[o]*tanh(Z+bias), softmax_j, agg = att@x (f32),
// h = agg@Wa + x_i@Wn + biases, BN(eval) + SELU, pool score = sigmoid(h@pw+pb).
// ---------------------------------------------------------------------------
__global__ __launch_bounds__(256)
void aasist_gat_kernel(const float* __restrict__ x,
                       const float* __restrict__ att_proj_w,
                       const float* __restrict__ att_proj_b,
                       const float* __restrict__ att_weight,
                       const float* __restrict__ proj_att_w,
                       const float* __restrict__ proj_att_b,
                       const float* __restrict__ proj_noatt_w,
                       const float* __restrict__ proj_noatt_b,
                       const float* __restrict__ bn_gamma,
                       const float* __restrict__ bn_beta,
                       const float* __restrict__ bn_mean,
                       const float* __restrict__ bn_var,
                       const float* __restrict__ pool_w,
                       const float* __restrict__ pool_b,
                       float* __restrict__ h_buf,
                       float* __restrict__ score_buf)
{
    __shared__ _Float16 xh[NN * XSTR];    // f16 copy of x[b]  (73.7 KB)
    __shared__ _Float16 wh[DD * OO];      // f16 att_proj_w    ( 8.2 KB)
    __shared__ float    att_s[8][NN];     // per-wave logits->att (16 KB)
    __shared__ float    aggs[8][DD];      // per-wave aggregate   ( 2 KB)

    const int tid = threadIdx.x;
    const int b   = blockIdx.x >> 6;          // 64 blocks per batch
    const int ib  = (blockIdx.x & 63) << 3;   // first of 8 i-rows
    const float* __restrict__ xb = x + (size_t)b * NN * DD;

    // Stage x[b] (f16) and att_proj_w (f16) into LDS.
    for (int t = tid; t < NN * DD; t += 256) {
        int j = t >> 6, d = t & 63;
        xh[j * XSTR + d] = (_Float16)xb[t];
    }
    for (int t = tid; t < DD * OO; t += 256)
        wh[t] = (_Float16)att_proj_w[t];
    __syncthreads();

    const int wave = tid >> 5;
    const int lane = tid & 31;
    const int half = lane >> 4;
    const int lr   = lane & 15;
    const int i    = ib + wave;

    // B fragments: W as 32x16 tiles, [kstep][otile].  B layout (ISA 7.12.2):
    // lane -> N = lane%16; element e -> K = 8*half + e + (e&8) (within 32-row k-slab).
    v16h bfrag[2][4];
    #pragma unroll
    for (int ks = 0; ks < 2; ++ks)
        #pragma unroll
        for (int n = 0; n < 4; ++n)
            #pragma unroll
            for (int e = 0; e < 16; ++e)
                bfrag[ks][n][e] = wh[(32 * ks + 8 * half + e + (e & 8)) * OO + (16 * n + lr)];

    float biasv[4], attwv[4];
    #pragma unroll
    for (int n = 0; n < 4; ++n) {
        int o = 16 * n + lr;
        biasv[n] = att_proj_b[o];
        attwv[n] = att_weight[o];
    }

    const _Float16* __restrict__ xi = &xh[i * XSTR];

    // ---- logits over all j, 16 rows per tile ----
    for (int m = 0; m < 32; ++m) {
        const _Float16* __restrict__ xj = &xh[(16 * m + lr) * XSTR];
        // A layout (ISA 7.12.2, 16-bit A 16x32): lane -> M = lane%16;
        // element e (r=e/2,t=e&1) -> K = (r<4?0:16) + 8*half + 2*(r&3) + t.
        v16h a0, a1;
        #pragma unroll
        for (int e = 0; e < 16; ++e) {
            int r  = e >> 1, t = e & 1;
            int kl = ((r & 4) << 2) + 8 * half + 2 * (r & 3) + t;
            a0[e] = xj[kl]      * xi[kl];        // k-step 0: d = kl
            a1[e] = xj[32 + kl] * xi[32 + kl];   // k-step 1: d = 32+kl
        }
        v8f acc[4] = {};
        #pragma unroll
        for (int n = 0; n < 4; ++n) {
            acc[n] = __builtin_amdgcn_wmma_f32_16x16x32_f16(
                false, a0, false, bfrag[0][n], (short)0, acc[n], false, false);
            acc[n] = __builtin_amdgcn_wmma_f32_16x16x32_f16(
                false, a1, false, bfrag[1][n], (short)0, acc[n], false, false);
        }
        // tanh + dot with att_weight; C/D layout: VGPR r -> row M = r + 8*half.
        float s[8] = {0.f,0.f,0.f,0.f,0.f,0.f,0.f,0.f};
        #pragma unroll
        for (int n = 0; n < 4; ++n)
            #pragma unroll
            for (int r = 0; r < 8; ++r)
                s[r] += attwv[n] * fast_tanh(acc[n][r] + biasv[n]);
        #pragma unroll
        for (int off = 8; off >= 1; off >>= 1)
            #pragma unroll
            for (int r = 0; r < 8; ++r)
                s[r] += __shfl_xor(s[r], off, 32);   // reduce within 16-lane group
        if (lr == 0) {
            #pragma unroll
            for (int r = 0; r < 8; ++r)
                att_s[wave][16 * m + 8 * half + r] = s[r] * TEMP_INV;
        }
    }

    // ---- wave-local softmax over j ----
    float mx = -3.0e38f;
    for (int t = lane; t < NN; t += 32) mx = fmaxf(mx, att_s[wave][t]);
    #pragma unroll
    for (int off = 16; off >= 1; off >>= 1) mx = fmaxf(mx, __shfl_xor(mx, off, 32));
    float sum = 0.f;
    for (int t = lane; t < NN; t += 32) {
        float e = __expf(att_s[wave][t] - mx);
        att_s[wave][t] = e;
        sum += e;
    }
    #pragma unroll
    for (int off = 16; off >= 1; off >>= 1) sum += __shfl_xor(sum, off, 32);
    const float inv = 1.f / sum;
    for (int t = lane; t < NN; t += 32) att_s[wave][t] *= inv;

    // ---- agg = att @ x  (full f32, x from global / L2) ----
    const int d0 = 2 * lane, d1 = d0 + 1;
    float ag0 = 0.f, ag1 = 0.f;
    for (int j = 0; j < NN; ++j) {
        float a = att_s[wave][j];
        ag0 += a * xb[j * DD + d0];
        ag1 += a * xb[j * DD + d1];
    }
    aggs[wave][d0] = ag0;
    aggs[wave][d1] = ag1;       // same-wave LDS RAW: in-order DS, no barrier

    // ---- h = agg@Wa + x_i@Wn + biases ; BN(eval) ; SELU ----
    const int o0 = d0, o1 = d1;
    float h0 = proj_att_b[o0] + proj_noatt_b[o0];
    float h1 = proj_att_b[o1] + proj_noatt_b[o1];
    for (int d = 0; d < DD; ++d) {
        float ag = aggs[wave][d];
        float xv = xb[i * DD + d];
        h0 += ag * proj_att_w[d * OO + o0] + xv * proj_noatt_w[d * OO + o0];
        h1 += ag * proj_att_w[d * OO + o1] + xv * proj_noatt_w[d * OO + o1];
    }
    h0 = (h0 - bn_mean[o0]) * rsqrtf(bn_var[o0] + BN_EPS_C) * bn_gamma[o0] + bn_beta[o0];
    h1 = (h1 - bn_mean[o1]) * rsqrtf(bn_var[o1] + BN_EPS_C) * bn_gamma[o1] + bn_beta[o1];
    h0 = SELU_SCALE * (h0 > 0.f ? h0 : SELU_ALPHA * expm1f(h0));
    h1 = SELU_SCALE * (h1 > 0.f ? h1 : SELU_ALPHA * expm1f(h1));

    float* __restrict__ hrow = h_buf + ((size_t)b * NN + i) * OO;
    hrow[o0] = h0;
    hrow[o1] = h1;

    // ---- pool score = sigmoid(h @ pool_w + pool_b) ----
    float p = h0 * pool_w[o0] + h1 * pool_w[o1];
    #pragma unroll
    for (int off = 16; off >= 1; off >>= 1) p += __shfl_xor(p, off, 32);
    if (lane == 0) {
        float z = p + pool_b[0];
        score_buf[b * NN + i] = 1.f / (1.f + __expf(-z));
    }
}

// ---------------------------------------------------------------------------
// Kernel 2: GraphPool top-k (k=256, sorted desc, ties -> lower index) + gather.
// One block per batch; bitonic sort of 512 (score, idx) pairs in LDS.
// ---------------------------------------------------------------------------
__global__ __launch_bounds__(256)
void aasist_pool_kernel(const float* __restrict__ h_buf,
                        const float* __restrict__ score_buf,
                        float* __restrict__ out)
{
    __shared__ float key[NN];
    __shared__ int   ind[NN];
    const int tid = threadIdx.x;
    const int b   = blockIdx.x;

    for (int t = tid; t < NN; t += 256) { key[t] = score_buf[b * NN + t]; ind[t] = t; }
    __syncthreads();

    for (int k = 2; k <= NN; k <<= 1) {
        for (int j = k >> 1; j > 0; j >>= 1) {
            for (int t = tid; t < NN; t += 256) {
                int p = t ^ j;
                if (p > t) {
                    float a = key[t], c = key[p];
                    int   ia = ind[t], ic = ind[p];
                    bool a_first = (a > c) || (a == c && ia < ic);  // desc, idx-asc ties
                    bool desc    = ((t & k) == 0);
                    if (desc ? !a_first : a_first) {
                        key[t] = c; key[p] = a;
                        ind[t] = ic; ind[p] = ia;
                    }
                }
            }
            __syncthreads();
        }
    }

    // out[b,k,:] = h[b, idx_k, :] * score[idx_k]
    for (int t = tid; t < NKP * OO; t += 256) {
        int kk = t >> 6, o = t & 63;
        int j  = ind[kk];
        out[((size_t)b * NKP + kk) * OO + o] =
            h_buf[((size_t)b * NN + j) * OO + o] * key[kk];
    }
}

// ---------------------------------------------------------------------------
extern "C" void kernel_launch(void* const* d_in, const int* in_sizes, int n_in,
                              void* d_out, int out_size, void* d_ws, size_t ws_size,
                              hipStream_t stream)
{
    (void)in_sizes; (void)n_in; (void)out_size; (void)ws_size;
    const float* x            = (const float*)d_in[0];
    const float* att_proj_w   = (const float*)d_in[1];
    const float* att_proj_b   = (const float*)d_in[2];
    const float* att_weight   = (const float*)d_in[3];
    const float* proj_att_w   = (const float*)d_in[4];
    const float* proj_att_b   = (const float*)d_in[5];
    const float* proj_noatt_w = (const float*)d_in[6];
    const float* proj_noatt_b = (const float*)d_in[7];
    const float* bn_gamma     = (const float*)d_in[8];
    const float* bn_beta      = (const float*)d_in[9];
    const float* bn_mean      = (const float*)d_in[10];
    const float* bn_var       = (const float*)d_in[11];
    const float* pool_w       = (const float*)d_in[12];
    const float* pool_b       = (const float*)d_in[13];

    float* h_buf     = (float*)d_ws;                                   // B*N*O f32 = 512 KB
    float* score_buf = (float*)((char*)d_ws +
                                (size_t)BB * NN * OO * sizeof(float)); // B*N f32 = 8 KB

    aasist_gat_kernel<<<BB * (NN / 8), 256, 0, stream>>>(
        x, att_proj_w, att_proj_b, att_weight,
        proj_att_w, proj_att_b, proj_noatt_w, proj_noatt_b,
        bn_gamma, bn_beta, bn_mean, bn_var, pool_w, pool_b,
        h_buf, score_buf);

    aasist_pool_kernel<<<BB, 256, 0, stream>>>(h_buf, score_buf, (float*)d_out);
}
